// HQCNN_27290222198805
// MI455X (gfx1250) — matrix operand order repaired and checked by reading, not compiled, not run on hardware
//
#include <hip/hip_runtime.h>
#include <hip/hip_bf16.h>

typedef __attribute__((ext_vector_type(16))) _Float16 v16h;
typedef __attribute__((ext_vector_type(8)))  float    v8f;

// ---------------------------------------------------------------------------
// Complex helpers (float2 = {re, im})
// ---------------------------------------------------------------------------
__device__ __forceinline__ float2 cmulf(float2 a, float2 b) {
    return make_float2(a.x * b.x - a.y * b.y, a.x * b.y + a.y * b.x);
}

__device__ void mm4(const float2* A, const float2* B, float2* C) {
    for (int i = 0; i < 4; ++i)
        for (int j = 0; j < 4; ++j) {
            float2 acc = make_float2(0.f, 0.f);
            for (int k = 0; k < 4; ++k) {
                float2 p = cmulf(A[i * 4 + k], B[k * 4 + j]);
                acc.x += p.x; acc.y += p.y;
            }
            C[i * 4 + j] = acc;
        }
}

__device__ void kron22(const float2* A, const float2* B, float2* K) {
    for (int i = 0; i < 2; ++i)
        for (int j = 0; j < 2; ++j)
            for (int k = 0; k < 2; ++k)
                for (int l = 0; l < 2; ++l)
                    K[(2 * i + k) * 4 + (2 * j + l)] = cmulf(A[i * 2 + j], B[k * 2 + l]);
}

__device__ void u3m(float t, float p, float d, float2* M) {
    float ct = cosf(0.5f * t), st = sinf(0.5f * t);
    M[0] = make_float2(ct, 0.f);
    M[1] = make_float2(-cosf(d) * st, -sinf(d) * st);
    M[2] = make_float2(cosf(p) * st, sinf(p) * st);
    M[3] = make_float2(cosf(p + d) * ct, sinf(p + d) * ct);
}
__device__ void rym(float t, float2* M) {
    float c = cosf(0.5f * t), s = sinf(0.5f * t);
    M[0] = make_float2(c, 0.f); M[1] = make_float2(-s, 0.f);
    M[2] = make_float2(s, 0.f); M[3] = make_float2(c, 0.f);
}
__device__ void rzm(float t, float2* M) {
    float c = cosf(0.5f * t), s = sinf(0.5f * t);
    M[0] = make_float2(c, -s); M[1] = make_float2(0.f, 0.f);
    M[2] = make_float2(0.f, 0.f); M[3] = make_float2(c, s);
}
__device__ void swaprows(float2* M, int r0, int r1) {
    for (int j = 0; j < 4; ++j) {
        float2 t = M[r0 * 4 + j]; M[r0 * 4 + j] = M[r1 * 4 + j]; M[r1 * 4 + j] = t;
    }
}

// ---------------------------------------------------------------------------
// Kernel 0: build the shared 4x4 SU(4) conv gate U and pool gate P (once).
// ---------------------------------------------------------------------------
__global__ void build_gates_kernel(const float* w0, const float* w1, const float* w2,
                                   const float* w3, const float* w4, const float* w5,
                                   const float* w6, const float* w7, const float* w8,
                                   float* gm) {
    if (threadIdx.x != 0) return;
    float2 A[4], Bm[4], K[16], U[16], T[16];
    u3m(w0[0], w0[1], w0[2], A);
    u3m(w1[0], w1[1], w1[2], Bm);
    kron22(A, Bm, U);
    swaprows(U, 2, 3);                              // CNOT01 @ U
    rym(w2[0], A); rzm(w3[0], Bm);
    kron22(A, Bm, K); mm4(K, U, T);                 // kron(ry,rz) @ U
    swaprows(T, 1, 3);                              // CNOT10 @ .
    rym(w4[0], A);
    Bm[0] = make_float2(1.f, 0.f); Bm[1] = make_float2(0.f, 0.f);
    Bm[2] = make_float2(0.f, 0.f); Bm[3] = make_float2(1.f, 0.f);
    kron22(A, Bm, K); mm4(K, T, U);                 // kron(ry,I) @ .
    swaprows(U, 2, 3);                              // CNOT01 @ .
    u3m(w5[0], w5[1], w5[2], A);
    u3m(w6[0], w6[1], w6[2], Bm);
    kron22(A, Bm, K); mm4(K, U, T);                 // kron(u3,u3) @ .
    float2* g = (float2*)gm;
    for (int k = 0; k < 16; ++k) g[k] = T[k];

    // Pool: P = CRX @ kron(X,I2) @ CRZ
    float2 CRZ[16], XI[16], CRX[16], M1[16], P[16];
    for (int k = 0; k < 16; ++k) {
        CRZ[k] = make_float2(0.f, 0.f); XI[k] = make_float2(0.f, 0.f);
        CRX[k] = make_float2(0.f, 0.f);
    }
    float t7 = w7[0];
    CRZ[0]  = make_float2(1.f, 0.f); CRZ[5] = make_float2(1.f, 0.f);
    CRZ[10] = make_float2(cosf(0.5f * t7), -sinf(0.5f * t7));
    CRZ[15] = make_float2(cosf(0.5f * t7),  sinf(0.5f * t7));
    XI[0 * 4 + 2] = make_float2(1.f, 0.f); XI[1 * 4 + 3] = make_float2(1.f, 0.f);
    XI[2 * 4 + 0] = make_float2(1.f, 0.f); XI[3 * 4 + 1] = make_float2(1.f, 0.f);
    float c8 = cosf(0.5f * w8[0]), s8 = sinf(0.5f * w8[0]);
    CRX[0]  = make_float2(1.f, 0.f); CRX[5] = make_float2(1.f, 0.f);
    CRX[10] = make_float2(c8, 0.f);  CRX[11] = make_float2(0.f, -s8);
    CRX[14] = make_float2(0.f, -s8); CRX[15] = make_float2(c8, 0.f);
    mm4(XI, CRZ, M1);
    mm4(CRX, M1, P);
    for (int k = 0; k < 16; ++k) g[16 + k] = P[k];
}

// ---------------------------------------------------------------------------
// Kernel 1: anglesT[8][B] = (x[B,64] @ W1^T + b1)^T via v_wmma_f32_16x16x32_f16.
// One wave per 16-row tile. All loads unconditional (mask-by-multiply), b128.
// ---------------------------------------------------------------------------
__device__ __forceinline__ void pack8(v16h& v, int base, float4 a, float4 b) {
    v[base + 0] = (_Float16)a.x; v[base + 1] = (_Float16)a.y;
    v[base + 2] = (_Float16)a.z; v[base + 3] = (_Float16)a.w;
    v[base + 4] = (_Float16)b.x; v[base + 5] = (_Float16)b.y;
    v[base + 6] = (_Float16)b.z; v[base + 7] = (_Float16)b.w;
}

__global__ void angles_wmma_kernel(const float* __restrict__ x,
                                   const float* __restrict__ W1,
                                   const float* __restrict__ b1,
                                   float* __restrict__ anglesT, int Bn) {
    int wave = blockIdx.x * (blockDim.x >> 5) + (threadIdx.x >> 5);
    int lane = threadIdx.x & 31;
    int m0   = wave * 16;
    if (m0 >= Bn) return;                 // wave-uniform; EXEC stays all-ones
    int half = lane >> 4;                 // 0: lanes 0-15, 1: lanes 16-31
    int col  = lane & 15;                 // A: M index; B/C: N index

    // ---- A operand: x rows, f32 -> f16. K chunks per ISA layout:
    // half 0 lanes hold K {kb..kb+7, kb+16..kb+23}, kb = half*8.
    const float4* xr = (const float4*)(x + (size_t)(m0 + col) * 64);
    int h2 = half * 2;                    // float4 index of kb
    float4 f0 = xr[h2 + 0],  f1 = xr[h2 + 1];    // K kb..kb+7
    float4 f2 = xr[h2 + 4],  f3 = xr[h2 + 5];    // K kb+16..kb+23
    float4 f4 = xr[h2 + 8],  f5 = xr[h2 + 9];    // K kb+32..kb+39
    float4 f6 = xr[h2 + 12], f7 = xr[h2 + 13];   // K kb+48..kb+55
    v16h a0, a1;
    pack8(a0, 0, f0, f1); pack8(a0, 8, f2, f3);
    pack8(a1, 0, f4, f5); pack8(a1, 8, f6, f7);

    // ---- B operand: B[k][n] = W1[n][k], n = col (cols 8..15 zeroed by mask).
    // half h lanes hold K {8h..8h+7, 16+8h..16+8h+7} (+32 for second WMMA).
    float wmask = (col < 8) ? 1.f : 0.f;
    const float4* wr = (const float4*)(W1 + (size_t)(col & 7) * 64);  // always in-bounds
    int wb = half * 2;
    float4 g0 = wr[wb + 0], g1 = wr[wb + 1], g2 = wr[wb + 4], g3 = wr[wb + 5];
    float4 g4 = wr[wb + 8], g5 = wr[wb + 9], g6 = wr[wb + 12], g7 = wr[wb + 13];
#define MSK4(v) v.x *= wmask; v.y *= wmask; v.z *= wmask; v.w *= wmask
    MSK4(g0); MSK4(g1); MSK4(g2); MSK4(g3);
    MSK4(g4); MSK4(g5); MSK4(g6); MSK4(g7);
#undef MSK4
    v16h b0, bm1;
    pack8(b0, 0, g0, g1); pack8(b0, 8, g2, g3);
    pack8(bm1, 0, g4, g5); pack8(bm1, 8, g6, g7);

    // ---- C init with bias, two chained WMMAs.
    float bias = b1[col & 7] * wmask;
    v8f c;
#pragma unroll
    for (int r = 0; r < 8; ++r) c[r] = bias;
    c = __builtin_amdgcn_wmma_f32_16x16x32_f16(false, a0, false, b0,  (short)0, c, false, false);
    c = __builtin_amdgcn_wmma_f32_16x16x32_f16(false, a1, false, bm1, (short)0, c, false, false);

    // ---- Store transposed: anglesT[col][m0 + half*8 + r], r = 0..7 contiguous.
    if (col < 8) {
        float* orow = anglesT + (size_t)col * Bn + m0 + half * 8;
        float4 o0 = make_float4(c[0], c[1], c[2], c[3]);
        float4 o1 = make_float4(c[4], c[5], c[6], c[7]);
        *(float4*)(orow + 0) = o0;
        *(float4*)(orow + 4) = o1;
    }
}

// ---------------------------------------------------------------------------
// Kernel 2: 8-qubit statevector sim, one wave32 per batch element.
// Amplitude index s = lane[4:0]:reg[2:0]; qubit w <-> bit w of s.
// ---------------------------------------------------------------------------
template <int F>
__device__ __forceinline__ float2 xfetch(const float2* st, int r) {
    constexpr int RP = F & 7;       // register-space flip
    constexpr int LP = F >> 3;      // lane-space flip
    float2 v = st[r ^ RP];
    if constexpr (LP != 0) {
        v.x = __shfl_xor(v.x, LP, 32);
        v.y = __shfl_xor(v.y, LP, 32);
    }
    return v;
}

// acc += M[i][i^T] * v, with i = qa*2+qb selected via cndmask chains.
template <int T>
__device__ __forceinline__ void cfma_sel(float2& acc, const float2* M,
                                         bool qa, bool qb, float2 v) {
    float2 u0 = M[0 * 4 + (0 ^ T)];
    float2 u1 = M[1 * 4 + (1 ^ T)];
    float2 u2 = M[2 * 4 + (2 ^ T)];
    float2 u3v = M[3 * 4 + (3 ^ T)];
    float cx0 = qb ? u1.x : u0.x, cy0 = qb ? u1.y : u0.y;
    float cx1 = qb ? u3v.x : u2.x, cy1 = qb ? u3v.y : u2.y;
    float cx = qa ? cx1 : cx0, cy = qa ? cy1 : cy0;
    acc.x += cx * v.x - cy * v.y;
    acc.y += cx * v.y + cy * v.x;
}

// Apply shared 4x4 gate M on qubits (PA, PB): new(s) = sum_t M[i][i^t]*amp(s^f_t)
template <int PA, int PB>
__device__ __forceinline__ void apply2(float2* st, const float2* M, int lane) {
    constexpr int ma = 1 << PA, mb = 1 << PB;
    float2 ns[8];
#pragma unroll
    for (int r = 0; r < 8; ++r) {
        int s = (lane << 3) | r;
        bool qa = (s & ma) != 0, qb = (s & mb) != 0;
        float2 v0 = st[r];
        float2 v1 = xfetch<mb>(st, r);
        float2 v2 = xfetch<ma>(st, r);
        float2 v3 = xfetch<ma | mb>(st, r);
        float2 acc = make_float2(0.f, 0.f);
        cfma_sel<0>(acc, M, qa, qb, v0);
        cfma_sel<1>(acc, M, qa, qb, v1);
        cfma_sel<2>(acc, M, qa, qb, v2);
        cfma_sel<3>(acc, M, qa, qb, v3);
        ns[r] = acc;
    }
#pragma unroll
    for (int r = 0; r < 8; ++r) st[r] = ns[r];
}

__global__ void circuit_kernel(const float* __restrict__ anglesT,
                               const float* __restrict__ gm,
                               const float* __restrict__ W2,
                               const float* __restrict__ b2,
                               float* __restrict__ out, int Bn) {
    int wave = blockIdx.x * (blockDim.x >> 5) + (threadIdx.x >> 5);
    int lane = threadIdx.x & 31;
    if (wave >= Bn) return;   // wave-uniform

    // Embedding: product state after RX(angle_w) on each qubit:
    // amp(s) = (-i)^popc(s) * prod_w (bit_w ? sin : cos)(a_w/2)
    float cw[8], sw[8];
#pragma unroll
    for (int w = 0; w < 8; ++w) {
        float h = 0.5f * anglesT[(size_t)w * Bn + wave];
        cw[w] = cosf(h); sw[w] = sinf(h);
    }
    float2 st[8];
#pragma unroll
    for (int r = 0; r < 8; ++r) {
        int s = (lane << 3) | r;
        float p = 1.f;
#pragma unroll
        for (int w = 0; w < 8; ++w)
            p *= ((s >> w) & 1) ? sw[w] : cw[w];
        int m = __popc(s) & 3;
        float re = (m == 0) ? p : ((m == 2) ? -p : 0.f);
        float im = (m == 1) ? -p : ((m == 3) ? p : 0.f);
        st[r] = make_float2(re, im);
    }

    // Load shared gate matrices into registers (wave-uniform values).
    float2 U[16], P[16];
    const float2* g = (const float2*)gm;
#pragma unroll
    for (int k = 0; k < 16; ++k) { U[k] = g[k]; P[k] = g[16 + k]; }

    // Conv layer: shared SU(4) on ring pairs.
    apply2<0, 1>(st, U, lane);
    apply2<2, 3>(st, U, lane);
    apply2<4, 5>(st, U, lane);
    apply2<6, 7>(st, U, lane);
    apply2<1, 2>(st, U, lane);
    apply2<3, 4>(st, U, lane);
    apply2<5, 6>(st, U, lane);
    apply2<7, 0>(st, U, lane);
    // Pool layer.
    apply2<0, 1>(st, P, lane);
    apply2<2, 3>(st, P, lane);
    apply2<4, 5>(st, P, lane);
    apply2<6, 7>(st, P, lane);

    // <Z> on wires 1,3,5,7 then final linear layer.
    float acc0 = 0.f, acc1 = 0.f, acc2 = 0.f, acc3 = 0.f;
#pragma unroll
    for (int r = 0; r < 8; ++r) {
        int s = (lane << 3) | r;
        float pr = st[r].x * st[r].x + st[r].y * st[r].y;
        acc0 += (s & (1 << 1)) ? -pr : pr;
        acc1 += (s & (1 << 3)) ? -pr : pr;
        acc2 += (s & (1 << 5)) ? -pr : pr;
        acc3 += (s & (1 << 7)) ? -pr : pr;
    }
#pragma unroll
    for (int off = 16; off >= 1; off >>= 1) {
        acc0 += __shfl_xor(acc0, off, 32);
        acc1 += __shfl_xor(acc1, off, 32);
        acc2 += __shfl_xor(acc2, off, 32);
        acc3 += __shfl_xor(acc3, off, 32);
    }
    if (lane == 0) {
        out[wave] = acc0 * W2[0] + acc1 * W2[1] + acc2 * W2[2] + acc3 * W2[3] + b2[0];
    }
}

// ---------------------------------------------------------------------------
// Launch: ws = [ anglesT: 8*B f32 | U: 16 float2 | P: 16 float2 ]  (~2 MB)
// ---------------------------------------------------------------------------
extern "C" void kernel_launch(void* const* d_in, const int* in_sizes, int n_in,
                              void* d_out, int out_size, void* d_ws, size_t ws_size,
                              hipStream_t stream) {
    (void)n_in; (void)out_size; (void)ws_size;
    const float* x  = (const float*)d_in[0];
    const float* W1 = (const float*)d_in[1];
    const float* b1 = (const float*)d_in[2];
    const float* w0 = (const float*)d_in[3];
    const float* w1 = (const float*)d_in[4];
    const float* w2 = (const float*)d_in[5];
    const float* w3 = (const float*)d_in[6];
    const float* w4 = (const float*)d_in[7];
    const float* w5 = (const float*)d_in[8];
    const float* w6 = (const float*)d_in[9];
    const float* w7 = (const float*)d_in[10];
    const float* w8 = (const float*)d_in[11];
    const float* W2 = (const float*)d_in[12];
    const float* b2 = (const float*)d_in[13];
    float* out = (float*)d_out;

    int Bn = in_sizes[0] / 64;                 // 65536
    float* ws      = (float*)d_ws;
    float* anglesT = ws;
    float* gm      = ws + (size_t)Bn * 8;

    build_gates_kernel<<<1, 1, 0, stream>>>(w0, w1, w2, w3, w4, w5, w6, w7, w8, gm);

    // One wave per 16-row tile, 8 waves (256 thr) per block.
    int tiles = Bn / 16;                       // 4096 waves
    angles_wmma_kernel<<<tiles / 8, 256, 0, stream>>>(x, W1, b1, anglesT, Bn);

    // One wave per batch element, 8 waves per block.
    circuit_kernel<<<Bn / 8, 256, 0, stream>>>(anglesT, gm, W2, b2, out, Bn);
}
